// Disentangled_Multi_Head_Attention_63539746177626
// MI455X (gfx1250) — matrix operand order repaired and checked by reading, not compiled, not run on hardware
//
#include <hip/hip_runtime.h>
#include <hip/hip_bf16.h>

// ---------------------------------------------------------------------------
// Disentangled MHA, algebraically folded so the two [B,S,S,H] time-embedding
// tensors are each streamed exactly ONCE (HBM-bound: ~655MB => ~28us @23.3TB/s)
// Dense GEMMs use v_wmma_f32_16x16x32_f16; the streaming LN-dot pass uses
// 8-lane-group reductions (3 butterfly stages) to stay above the HBM roofline.
// ---------------------------------------------------------------------------

#define Bb_ 8
#define S_ 200
#define H_ 256
#define NH_ 4
#define D_ 64
#define EPS_ 1e-12f

typedef __attribute__((ext_vector_type(16))) _Float16 v16h;
typedef __attribute__((ext_vector_type(8)))  float    v8f;
typedef __attribute__((ext_vector_type(4)))  float    fx4;

// ---- WMMA fragment loaders (exact CDNA5 ISA layouts, all loads contiguous) --
// A 16x32 f16: lane=(g=l>>4, r=l&15). VGPR0..3 <- K=8g..8g+7, VGPR4..7 <- K=16+8g..
__device__ __forceinline__ v16h frag_a(const _Float16* rowp, int g) {
  union { fx4 f[2]; v16h h; } u;
  u.f[0] = *(const fx4*)(rowp + 8 * g);
  u.f[1] = *(const fx4*)(rowp + 16 + 8 * g);
  return u.h;
}
// B 32x16 f16 with B[k,n] = Wt[n,k]: lane=(g,n). K=8g..8g+7 then K=16+8g.. from Wt row n.
__device__ __forceinline__ v16h frag_b(const _Float16* rowp, int g) {
  union { fx4 f[2]; v16h h; } u;
  u.f[0] = *(const fx4*)(rowp + 8 * g);
  u.f[1] = *(const fx4*)(rowp + 16 + 8 * g);
  return u.h;
}
__device__ __forceinline__ v16h frag_zero() {
  v16h z;
#pragma unroll
  for (int i = 0; i < 16; ++i) z[i] = (_Float16)0.f;
  return z;
}
__device__ __forceinline__ float wred32(float v) {
#pragma unroll
  for (int m = 16; m > 0; m >>= 1) v += __shfl_xor(v, m, 32);
  return v;
}
// reduce within aligned 8-lane groups (3 stages instead of 5)
__device__ __forceinline__ float wred8(float v) {
  v += __shfl_xor(v, 1, 32);
  v += __shfl_xor(v, 2, 32);
  v += __shfl_xor(v, 4, 32);
  return v;
}

// ---- fp32 -> f16 weight conversion: W[7,H,H] ++ dense_W[H,H] -----------------
__global__ void convert_wh(const float* __restrict__ W, const float* __restrict__ dW,
                           _Float16* __restrict__ Wh) {
  size_t i = (size_t)blockIdx.x * 256 + threadIdx.x;
  float v = (i < (size_t)7 * H_ * H_) ? W[i] : dW[i - (size_t)7 * H_ * H_];
  Wh[i] = (_Float16)v;
}

// ---- LayerNorm over rows of 256, output f16 --------------------------------
__global__ void ln_rows(const float* __restrict__ X, const float* __restrict__ g,
                        const float* __restrict__ be, _Float16* __restrict__ Y) {
  int r = blockIdx.x, t = threadIdx.x;
  float x = X[(size_t)r * H_ + t];
  __shared__ float red[256];
  red[t] = x; __syncthreads();
  for (int s = 128; s > 0; s >>= 1) { if (t < s) red[t] += red[t + s]; __syncthreads(); }
  float m = red[0] * (1.f / H_); __syncthreads();
  float dx = x - m;
  red[t] = dx * dx; __syncthreads();
  for (int s = 128; s > 0; s >>= 1) { if (t < s) red[t] += red[t + s]; __syncthreads(); }
  float rstd = rsqrtf(red[0] * (1.f / H_) + EPS_);
  Y[(size_t)r * H_ + t] = (_Float16)(dx * rstd * g[t] + be[t]);
}

// ---- generic WMMA GEMM: Y[M,N] = X[M,K] @ Wt[N,K]^T (+bias)(+residual) ------
__global__ void gemm_xwt(const _Float16* __restrict__ X, const _Float16* __restrict__ Wt,
                         const float* __restrict__ bias, const float* __restrict__ residual,
                         float* __restrict__ Y, int M, int N, int K) {
  int wave = threadIdx.x >> 5, lane = threadIdx.x & 31;
  int ntj = N >> 4;
  int tile = blockIdx.x * 4 + wave;
  if (tile >= (M >> 4) * ntj) return;         // uniform per wave
  int ti = tile / ntj, tj = tile - ti * ntj;
  int r = lane & 15, g = lane >> 4;
  const _Float16* Ab = X + ((size_t)(ti * 16 + r)) * K;
  const _Float16* Bp = Wt + ((size_t)(tj * 16 + r)) * K;
  v8f c;
#pragma unroll
  for (int q = 0; q < 8; ++q) c[q] = 0.f;
  for (int k = 0; k < K; k += 32) {
    v16h a = frag_a(Ab + k, g);
    v16h bv = frag_b(Bp + k, g);
    c = __builtin_amdgcn_wmma_f32_16x16x32_f16(false, a, false, bv, (short)0, c, false, false);
  }
  int col = tj * 16 + (lane & 15);
#pragma unroll
  for (int q = 0; q < 8; ++q) {
    int row = ti * 16 + g * 8 + q;
    float v = c[q];
    if (bias) v += bias[col];
    if (residual) v += residual[(size_t)row * N + col];
    Y[(size_t)row * N + col] = v;
  }
}

// ---- qp/kp f16 + V head-transpose (for contiguous B-frags in probs@V) -------
__global__ void fuse_qkv(const float* __restrict__ Q, const float* __restrict__ pQ,
                         const float* __restrict__ K, const float* __restrict__ pK,
                         const float* __restrict__ V, _Float16* __restrict__ qp,
                         _Float16* __restrict__ kp, _Float16* __restrict__ Vt) {
  int bi = blockIdx.x, t = threadIdx.x;
  size_t idx = (size_t)bi * H_ + t;
  qp[idx] = (_Float16)(Q[idx] + pQ[idx]);
  kp[idx] = (_Float16)(K[idx] + pK[idx]);
  int b = bi / S_, j = bi - b * S_;
  int h = t >> 6, n = t & 63;
  Vt[(((size_t)(b * NH_ + h) * D_) + n) * 224 + j] = (_Float16)V[idx];
}

// ---- w_tK[b,i,h,:] = W6_h^T @ (Q+pQ)_h ; c_tK = b6_h . (Q+pQ)_h -------------
__global__ void matvec_wtk(const float* __restrict__ Qf, const float* __restrict__ pQf,
                           const float* __restrict__ W6, const float* __restrict__ b6,
                           float* __restrict__ w_tk, float* __restrict__ c_tk) {
  int bih = blockIdx.x, bi = bih >> 2, h = bih & 3, t = threadIdx.x;
  __shared__ float u[64];
  if (t < 64) {
    size_t q = (size_t)bi * H_ + h * D_ + t;
    u[t] = Qf[q] + pQf[q];
  }
  __syncthreads();
  float acc = 0.f;
#pragma unroll 8
  for (int k = 0; k < 64; ++k) acc += u[k] * W6[(size_t)(h * D_ + k) * H_ + t];
  w_tk[(size_t)bih * H_ + t] = acc;
  if (t == 0) {
    float c = 0.f;
    for (int k = 0; k < 64; ++k) c += u[k] * b6[h * D_ + k];
    c_tk[bih] = c;
  }
}

// ---- single-pass stream over a [B,S,S,H] tensor -----------------------------
// Each wave handles 4 rows: lane = (sub = lane>>3 -> row, l8 = lane&7 -> 32-ch block).
// Cross-lane stats/dots reduce in 3 butterfly stages within 8-lane groups.
// mode 0 (tK): scores[b,h,i,j]  = LN(row).w_tk[b,i,h]; also accumulate mean_j LN(row)
// mode 1 (tQ): scores[b,h,i,j] += LN(row).w_tq[b,i,h] + csum[b,i,h]
__global__ void pass_t(const float* __restrict__ emb, const float* __restrict__ wv,
                       const float* __restrict__ gamma, const float* __restrict__ beta,
                       float* __restrict__ scores, float* __restrict__ mean_out,
                       const float* __restrict__ cadd, int mode) {
  int bi = blockIdx.x;
  int b = bi / S_, i = bi - b * S_;
  int t = threadIdx.x;
  int wave = t >> 5, lane = t & 31;
  int sub = lane >> 3, l8 = lane & 7;
  int c0 = l8 * 32;                       // this lane's 32 contiguous channels
  __shared__ float gw[4][256], wsh[4][256], gvec[256], bvec[256], meanacc[256];
  __shared__ float sgw[4], bw[4];
  gvec[t] = gamma[t]; bvec[t] = beta[t]; meanacc[t] = 0.f;
#pragma unroll
  for (int h = 0; h < 4; ++h) {
    float w = wv[((size_t)bi * 4 + h) * H_ + t];
    wsh[h][t] = w; gw[h][t] = gamma[t] * w;
  }
  __syncthreads();
  if (wave < 4) {
    int h = wave; float p1 = 0.f, p2 = 0.f;
#pragma unroll
    for (int e = 0; e < 8; ++e) { int c = lane * 8 + e; p1 += gw[h][c]; p2 += bvec[c] * wsh[h][c]; }
    p1 = wred32(p1); p2 = wred32(p2);
    if (lane == 0) { sgw[h] = p1; bw[h] = p2; }
  }
  __syncthreads();

  float acc[32];
  if (mode == 0) {
#pragma unroll
    for (int e = 0; e < 32; ++e) acc[e] = 0.f;
  }

  for (int jb = 0; jb < S_; jb += 32) {   // 8 waves x 4 rows per step
    int j = jb + wave * 4 + sub;
    bool valid = (j < S_);
    const float* row = emb + ((size_t)bi * S_ + (valid ? j : 0)) * H_ + c0;
    // prefetch next block's cachelines while this one computes
    if (jb + 32 < S_ && (j + 32) < S_)
      __builtin_prefetch(row + (size_t)32 * H_, 0, 0);
    float x[32];
    if (valid) {
#pragma unroll
      for (int ch = 0; ch < 8; ++ch) {
        fx4 v = *(const fx4*)(row + ch * 4);
        x[ch * 4 + 0] = v[0]; x[ch * 4 + 1] = v[1];
        x[ch * 4 + 2] = v[2]; x[ch * 4 + 3] = v[3];
      }
    } else {
#pragma unroll
      for (int e = 0; e < 32; ++e) x[e] = 0.f;
    }
    float s = 0.f, q = 0.f, d0 = 0.f, d1 = 0.f, d2 = 0.f, d3 = 0.f;
#pragma unroll
    for (int e = 0; e < 32; ++e) {
      float xe = x[e]; int c = c0 + e;
      s += xe; q = fmaf(xe, xe, q);
      d0 = fmaf(xe, gw[0][c], d0); d1 = fmaf(xe, gw[1][c], d1);
      d2 = fmaf(xe, gw[2][c], d2); d3 = fmaf(xe, gw[3][c], d3);
    }
    s = wred8(s); q = wred8(q);
    d0 = wred8(d0); d1 = wred8(d1); d2 = wred8(d2); d3 = wred8(d3);
    float mv = s * (1.f / H_);
    float rstd = rsqrtf(q * (1.f / H_) - mv * mv + EPS_);
    if (valid && l8 < 4) {
      float dh = (l8 == 0) ? d0 : (l8 == 1) ? d1 : (l8 == 2) ? d2 : d3;
      float val = rstd * (dh - mv * sgw[l8]) + bw[l8];
      size_t sidx = (((size_t)(b * NH_ + l8) * S_) + i) * S_ + j;
      if (mode == 0) scores[sidx] = val;
      else           scores[sidx] += val + cadd[bi * 4 + l8];
    }
    if (mode == 0 && valid) {
#pragma unroll
      for (int e = 0; e < 32; ++e) {
        int c = c0 + e;
        acc[e] += fmaf((x[e] - mv) * rstd, gvec[c], bvec[c]);
      }
    }
  }
  if (mode == 0) {
    __syncthreads();
#pragma unroll
    for (int e = 0; e < 32; ++e) atomicAdd(&meanacc[c0 + e], acc[e]);
    __syncthreads();
    mean_out[(size_t)bi * H_ + t] = meanacc[t] * (1.f / S_);
  }
}

// ---- u = lin(tKn_mean,6)_h + (K+pK)_h ; w_tQ = W5_h^T u ; csum = b5_h.u + c_tK
__global__ void make_wtq(const float* __restrict__ W, const float* __restrict__ bfull,
                         const float* __restrict__ tkn_mean, const float* __restrict__ Kf,
                         const float* __restrict__ pKf, const float* __restrict__ c_tk,
                         float* __restrict__ w_tq, float* __restrict__ csum) {
  int bih = blockIdx.x, bi = bih >> 2, h = bih & 3, t = threadIdx.x;
  __shared__ float u[64];
  const float* W6 = W + (size_t)6 * H_ * H_;
  const float* W5 = W + (size_t)5 * H_ * H_;
  const float* b6 = bfull + 6 * H_;
  const float* b5 = bfull + 5 * H_;
  if (t < 64) {
    float a = b6[h * D_ + t];
    const float* wr = W6 + (size_t)(h * D_ + t) * H_;
    const float* tm = tkn_mean + (size_t)bi * H_;
#pragma unroll 8
    for (int c = 0; c < H_; ++c) a += wr[c] * tm[c];
    size_t q = (size_t)bi * H_ + h * D_ + t;
    u[t] = a + Kf[q] + pKf[q];
  }
  __syncthreads();
  float acc = 0.f;
#pragma unroll 8
  for (int k = 0; k < 64; ++k) acc += u[k] * W5[(size_t)(h * D_ + k) * H_ + t];
  w_tq[(size_t)bih * H_ + t] = acc;
  if (t == 0) {
    float c = 0.f;
    for (int k = 0; k < 64; ++k) c += u[k] * b5[h * D_ + k];
    csum[bih] = c + c_tk[bih];
  }
}

// ---- scores += (Q+pQ)_h @ (K+pK)_h^T  (per b,h; WMMA, 13x13 tiles of 16) ----
__global__ void score_gemm(const _Float16* __restrict__ qp, const _Float16* __restrict__ kp,
                           float* __restrict__ scores) {
  int bh = blockIdx.y, b = bh >> 2, h = bh & 3;
  int wave = threadIdx.x >> 5, lane = threadIdx.x & 31;
  int tile = blockIdx.x * 4 + wave;
  if (tile >= 169) return;
  int ti = tile / 13, tj = tile - ti * 13;
  int r = lane & 15, g = lane >> 4;
  int ai = ti * 16 + r, bn = tj * 16 + r;
  const _Float16* Ab = qp + ((size_t)(b * S_ + ai)) * H_ + h * D_;
  const _Float16* Bp = kp + ((size_t)(b * S_ + bn)) * H_ + h * D_;
  int n0 = lane & 15;
  v8f c;
#pragma unroll
  for (int q = 0; q < 8; ++q) {
    int row = ti * 16 + g * 8 + q, col = tj * 16 + n0;
    c[q] = (row < S_ && col < S_) ? scores[((size_t)bh * S_ + row) * S_ + col] : 0.f;
  }
#pragma unroll
  for (int k = 0; k < 64; k += 32) {
    v16h a = (ai < S_) ? frag_a(Ab + k, g) : frag_zero();
    v16h bv = (bn < S_) ? frag_b(Bp + k, g) : frag_zero();
    c = __builtin_amdgcn_wmma_f32_16x16x32_f16(false, a, false, bv, (short)0, c, false, false);
  }
#pragma unroll
  for (int q = 0; q < 8; ++q) {
    int row = ti * 16 + g * 8 + q, col = tj * 16 + n0;
    if (row < S_ && col < S_) scores[((size_t)bh * S_ + row) * S_ + col] = c[q];
  }
}

// ---- softmax over j with mask; emit f16 probs padded to 224 -----------------
__global__ void softmax_k(const float* __restrict__ scores, const float* __restrict__ mask,
                          _Float16* __restrict__ probs) {
  int row = blockIdx.x;                  // bh*S + i
  int bh = row / S_, i = row - bh * S_, b = bh >> 2;
  int t = threadIdx.x;
  __shared__ float red[256];
  float s = -3.4e38f;
  if (t < S_) s = scores[(size_t)row * S_ + t] * 0.125f + mask[((size_t)b * S_ + i) * S_ + t];
  red[t] = s; __syncthreads();
  for (int k = 128; k > 0; k >>= 1) { if (t < k) red[t] = fmaxf(red[t], red[t + k]); __syncthreads(); }
  float mx = red[0]; __syncthreads();
  float e = (t < S_) ? __expf(s - mx) : 0.f;
  red[t] = e; __syncthreads();
  for (int k = 128; k > 0; k >>= 1) { if (t < k) red[t] += red[t + k]; __syncthreads(); }
  float inv = 1.f / red[0];
  if (t < 224) probs[(size_t)row * 224 + t] = (_Float16)((t < S_) ? e * inv : 0.f);
}

// ---- ctx[b,i,h*64+n] = probs @ V_h  (WMMA; K padded to 224) -----------------
__global__ void ctx_gemm(const _Float16* __restrict__ probs, const _Float16* __restrict__ Vt,
                         float* __restrict__ ctx) {
  int bh = blockIdx.y, b = bh >> 2, h = bh & 3;
  int wave = threadIdx.x >> 5, lane = threadIdx.x & 31;
  int tile = blockIdx.x * 4 + wave;
  if (tile >= 52) return;
  int ti = tile / 4, tj = tile - ti * 4;
  int r = lane & 15, g = lane >> 4;
  int ai = ti * 16 + r;
  const _Float16* Ab = probs + ((size_t)bh * S_ + ai) * 224;
  const _Float16* Bp = Vt + ((size_t)bh * D_ + tj * 16 + r) * 224;
  v8f c;
#pragma unroll
  for (int q = 0; q < 8; ++q) c[q] = 0.f;
  for (int k = 0; k < 224; k += 32) {
    v16h a = (ai < S_) ? frag_a(Ab + k, g) : frag_zero();
    v16h bv = frag_b(Bp + k, g);
    c = __builtin_amdgcn_wmma_f32_16x16x32_f16(false, a, false, bv, (short)0, c, false, false);
  }
  int n0 = lane & 15;
#pragma unroll
  for (int q = 0; q < 8; ++q) {
    int row = ti * 16 + g * 8 + q;
    if (row < S_) ctx[((size_t)(b * S_ + row)) * H_ + h * D_ + tj * 16 + n0] = c[q];
  }
}

// ============================================================================
extern "C" void kernel_launch(void* const* d_in, const int* in_sizes, int n_in,
                              void* d_out, int out_size, void* d_ws, size_t ws_size,
                              hipStream_t stream) {
  const float* input  = (const float*)d_in[0];
  const float* mask   = (const float*)d_in[1];
  const float* posQ   = (const float*)d_in[2];
  const float* posK   = (const float*)d_in[3];
  const float* tQ_emb = (const float*)d_in[4];
  const float* tK_emb = (const float*)d_in[5];
  const float* gam    = (const float*)d_in[6];   // [6,256]
  const float* bet    = (const float*)d_in[7];
  const float* W      = (const float*)d_in[8];   // [7,256,256]
  const float* bias   = (const float*)d_in[9];   // [7,256]
  const float* dW     = (const float*)d_in[10];
  const float* db     = (const float*)d_in[11];
  float* out = (float*)d_out;

  char* ws = (char*)d_ws;
  size_t off = 0;
  auto alloc = [&](size_t bytes) { size_t r = off; off += (bytes + 511) & ~(size_t)511; return r; };
  const size_t ROWS = (size_t)Bb_ * S_;          // 1600
  _Float16* Wh   = (_Float16*)(ws + alloc((size_t)8 * H_ * H_ * 2));
  _Float16* xnh  = (_Float16*)(ws + alloc(ROWS * H_ * 2));
  _Float16* pqnh = (_Float16*)(ws + alloc(ROWS * H_ * 2));
  _Float16* pknh = (_Float16*)(ws + alloc(ROWS * H_ * 2));
  _Float16* cxnh = (_Float16*)(ws + alloc(ROWS * H_ * 2));
  float* Qf  = (float*)(ws + alloc(ROWS * H_ * 4));
  float* Kf  = (float*)(ws + alloc(ROWS * H_ * 4));
  float* Vf  = (float*)(ws + alloc(ROWS * H_ * 4));
  float* pQf = (float*)(ws + alloc(ROWS * H_ * 4));
  float* pKf = (float*)(ws + alloc(ROWS * H_ * 4));
  _Float16* qph = (_Float16*)(ws + alloc(ROWS * H_ * 2));
  _Float16* kph = (_Float16*)(ws + alloc(ROWS * H_ * 2));
  _Float16* Vth = (_Float16*)(ws + alloc((size_t)Bb_ * NH_ * D_ * 224 * 2));
  float* w_tk = (float*)(ws + alloc(ROWS * 4 * H_ * 4));
  float* w_tq = (float*)(ws + alloc(ROWS * 4 * H_ * 4));
  float* c_tk = (float*)(ws + alloc(ROWS * 4 * 4));
  float* csum = (float*)(ws + alloc(ROWS * 4 * 4));
  float* tknm = (float*)(ws + alloc(ROWS * H_ * 4));
  float* scores = (float*)(ws + alloc((size_t)Bb_ * NH_ * S_ * S_ * 4));
  _Float16* probs = (_Float16*)(ws + alloc((size_t)Bb_ * NH_ * S_ * 224 * 2));
  float* ctx = (float*)(ws + alloc(ROWS * H_ * 4));
  (void)ws_size; (void)in_sizes; (void)n_in; (void)out_size;

  // 1) weights fp32->f16 (W0..W6 ++ dense_W)
  convert_wh<<<2048, 256, 0, stream>>>(W, dW, Wh);
  // 2) LayerNorms of the three [B,S,H] inputs
  ln_rows<<<ROWS, 256, 0, stream>>>(input, gam + 0 * H_, bet + 0 * H_, xnh);
  ln_rows<<<ROWS, 256, 0, stream>>>(posQ,  gam + 1 * H_, bet + 1 * H_, pqnh);
  ln_rows<<<ROWS, 256, 0, stream>>>(posK,  gam + 2 * H_, bet + 2 * H_, pknh);
  // 3) five projections (M=1600,N=256,K=256): 400 blocks x 4 waves
  gemm_xwt<<<400, 128, 0, stream>>>(xnh,  Wh + (size_t)0 * H_ * H_, bias + 0 * H_, nullptr, Qf,  1600, 256, 256);
  gemm_xwt<<<400, 128, 0, stream>>>(xnh,  Wh + (size_t)1 * H_ * H_, bias + 1 * H_, nullptr, Kf,  1600, 256, 256);
  gemm_xwt<<<400, 128, 0, stream>>>(xnh,  Wh + (size_t)2 * H_ * H_, bias + 2 * H_, nullptr, Vf,  1600, 256, 256);
  gemm_xwt<<<400, 128, 0, stream>>>(pqnh, Wh + (size_t)3 * H_ * H_, bias + 3 * H_, nullptr, pQf, 1600, 256, 256);
  gemm_xwt<<<400, 128, 0, stream>>>(pknh, Wh + (size_t)4 * H_ * H_, bias + 4 * H_, nullptr, pKf, 1600, 256, 256);
  // 4) qp/kp f16, V head-transpose
  fuse_qkv<<<ROWS, 256, 0, stream>>>(Qf, pQf, Kf, pKf, Vf, qph, kph, Vth);
  // 5) fold (Q+pQ) through W6: w_tK, c_tK
  matvec_wtk<<<ROWS * 4, 256, 0, stream>>>(Qf, pQf, W + (size_t)6 * H_ * H_, bias + 6 * H_, w_tk, c_tk);
  // 6) single pass over tK_emb: score term 3+7 and mean_j LN(tK_emb)
  pass_t<<<ROWS, 256, 0, stream>>>(tK_emb, w_tk, gam + 4 * H_, bet + 4 * H_, scores, tknm, nullptr, 0);
  // 7) u = tK_mean_h + (K+pK)_h folded through W5: w_tQ, csum
  make_wtq<<<ROWS * 4, 256, 0, stream>>>(W, bias, tknm, Kf, pKf, c_tk, w_tq, csum);
  // 8) single pass over tQ_emb: score terms 4+5+6 (+bias constants)
  pass_t<<<ROWS, 256, 0, stream>>>(tQ_emb, w_tq, gam + 3 * H_, bet + 3 * H_, scores, nullptr, csum, 1);
  // 9) scores += (Q+pQ)(K+pK)^T per (b,h)
  score_gemm<<<dim3(43, 32), 128, 0, stream>>>(qph, kph, scores);
  // 10) softmax (scale 1/8, +mask)
  softmax_k<<<(int)(ROWS * 4), 256, 0, stream>>>(scores, mask, probs);
  // 11) ctx = probs @ V per (b,h)
  ctx_gemm<<<dim3(13, 32), 128, 0, stream>>>(probs, Vth, ctx);
  // 12) LN(ctx) then dense + residual -> out
  ln_rows<<<ROWS, 256, 0, stream>>>(ctx, gam + 5 * H_, bet + 5 * H_, cxnh);
  gemm_xwt<<<400, 128, 0, stream>>>(cxnh, Wh + (size_t)7 * H_ * H_, db, input, out, 1600, 256, 256);
}
